// DynamicHybridModulation_27805618274416
// MI455X (gfx1250) — matrix operand order repaired and checked by reading, not compiled, not run on hardware
//
#include <hip/hip_runtime.h>
#include <hip/hip_bf16.h>

// MI455X (gfx1250) implementation of DynamicHybridModulation.
// All three GEMM stages (QKV projection, QK^T scores, probs@V context) run on
// V_WMMA_F32_16X16X32_F16 (wave32, f16 in / f32 accumulate). Wave tiles are
// sized so each staged LDS tile feeds many WMMAs (qkv: 8 wmma per wave per
// K-step). Glue math (spike indicators, t row/col means, squeeze+BN, gates,
// softmax) is VALU and negligible (~21 GFLOP total, <200 MB traffic vs
// 23.3 TB/s HBM -> matrix-pipe bound).

typedef __attribute__((ext_vector_type(16))) _Float16 v16h;
typedef __attribute__((ext_vector_type(8)))  float    v8f;

#define LDP 40  // padded LDS k-stride (halfs) for 32-wide K tiles

// ---------------------------------------------------------------------------
// WMMA fragment loaders (ISA 7.12.2 layouts, wave32)
// A-matrix 16x32 f16: lane -> row M = lane&15; VGPR v holds K pair with the
// documented half-swap: K = 2v + 8*(lane>>4) + (v>=4 ? 8 : 0).
__device__ __forceinline__ v16h frag_ld_a(const _Float16* base) {
  const int lane = threadIdx.x & 31;
  const int m  = lane & 15;
  const int lg = lane >> 4;
  const _Float16* row = base + m * LDP;
  v16h f;
#pragma unroll
  for (int v = 0; v < 8; ++v) {
    const int kb = 2 * v + 8 * lg + ((v >= 4) ? 8 : 0);
    f[2 * v]     = row[kb];
    f[2 * v + 1] = row[kb + 1];
  }
  return f;
}
// B-matrix 32x16 f16: lane -> col N = lane&15; same K order as A:
// K = 2v + 8*(lane>>4) + (v>=4 ? 8 : 0). LDS tile staged as [n][k]
// so each VGPR pair is one aligned 32-bit LDS read (fuses to ds_load_b128 x2).
__device__ __forceinline__ v16h frag_ld_b(const _Float16* base) {
  const int lane = threadIdx.x & 31;
  const int n  = lane & 15;
  const int lg = lane >> 4;
  const _Float16* row = base + n * LDP;
  v16h f;
#pragma unroll
  for (int v = 0; v < 8; ++v) {
    const int kb = 2 * v + 8 * lg + ((v >= 4) ? 8 : 0);
    f[2 * v]     = row[kb];
    f[2 * v + 1] = row[kb + 1];
  }
  return f;
}

__device__ __forceinline__ v8f wmma32(v16h a, v16h b, v8f c) {
  return __builtin_amdgcn_wmma_f32_16x16x32_f16(false, a, false, b,
                                                (short)0, c, false, false);
}

// C/D f32 16x16 layout per lane: n = lane&15, m = r + 8*(lane>>4).

// ---------------------------------------------------------------------------
__global__ __launch_bounds__(256) void k_zero(float* p, int n) {
  int i = blockIdx.x * 256 + threadIdx.x;
  if (i < n) p[i] = 0.f;
}

// ---------------------------------------------------------------------------
// Fused QKV projection: [4096,768] x [768,768] per z in {q,k,v}.
// Block tile 128x128, 8 waves, wave tile 32x64 (2 A-frags x 4 B-frags
// -> 8 WMMA per K-step). Activation fused in epilogue, outputs f16.
__global__ __launch_bounds__(256) void k_qkv(
    const float* __restrict__ X,
    const float* __restrict__ Wq, const float* __restrict__ Wk,
    const float* __restrict__ Wv,
    const float* __restrict__ bq, const float* __restrict__ bk,
    const float* __restrict__ bv,
    _Float16* __restrict__ qh, _Float16* __restrict__ kh,
    _Float16* __restrict__ vh) {
  const int which = blockIdx.z;
  const float* W    = which == 0 ? Wq : (which == 1 ? Wk : Wv);
  const float* bias = which == 0 ? bq : (which == 1 ? bk : bv);
  _Float16*    out  = which == 0 ? qh : (which == 1 ? kh : vh);

  const int m0 = blockIdx.x * 128;
  const int n0 = blockIdx.y * 128;
  const int t  = threadIdx.x;
  const int wave = t >> 5;
  const int wm = wave >> 1;  // 0..3 : 32-row strip
  const int wn = wave & 1;   // 0..1 : 64-col strip

  __shared__ _Float16 Xs[128 * LDP];
  __shared__ _Float16 Ws[128 * LDP];

  v8f acc[2][4];
#pragma unroll
  for (int i = 0; i < 2; ++i)
#pragma unroll
    for (int j = 0; j < 4; ++j) acc[i][j] = (v8f){};

  for (int k0 = 0; k0 < 768; k0 += 32) {
    __syncthreads();
    {  // stage X tile 128x32 (f32 -> f16), coalesced along k
      const int kk = t & 31, mm = t >> 5;
      if (k0 + 32 < 768)
        __builtin_prefetch(&X[(size_t)(m0 + mm) * 768 + k0 + 32], 0, 1);
#pragma unroll
      for (int p = 0; p < 16; ++p) {
        const int row = mm + p * 8;
        Xs[row * LDP + kk] = (_Float16)X[(size_t)(m0 + row) * 768 + k0 + kk];
      }
    }
    {  // stage W tile transposed: Ws[n][k] = W[(k0+k)*768 + n0+n]
      const int nn = t & 127, kk0 = t >> 7;
#pragma unroll
      for (int p = 0; p < 16; ++p) {
        const int kk = kk0 + p * 2;
        Ws[nn * LDP + kk] = (_Float16)W[(size_t)(k0 + kk) * 768 + n0 + nn];
      }
    }
    __syncthreads();
    v16h a[2], b[4];
#pragma unroll
    for (int mi = 0; mi < 2; ++mi)
      a[mi] = frag_ld_a(Xs + (wm * 32 + mi * 16) * LDP);
#pragma unroll
    for (int ni = 0; ni < 4; ++ni)
      b[ni] = frag_ld_b(Ws + (wn * 64 + ni * 16) * LDP);
#pragma unroll
    for (int mi = 0; mi < 2; ++mi)
#pragma unroll
      for (int ni = 0; ni < 4; ++ni)
        acc[mi][ni] = wmma32(a[mi], b[ni], acc[mi][ni]);
  }

  const int lane = t & 31;
  const int nloc = lane & 15, mh = lane >> 4;
#pragma unroll
  for (int mi = 0; mi < 2; ++mi) {
#pragma unroll
    for (int ni = 0; ni < 4; ++ni) {
#pragma unroll
      for (int r = 0; r < 8; ++r) {
        const int m = wm * 32 + mi * 16 + r + 8 * mh;
        const int n = wn * 64 + ni * 16 + nloc;
        float x = acc[mi][ni][r] + bias[n0 + n];
        float spike = (x >= 1.f) ? 1.f : ((x <= -1.f) ? -1.f : 0.f);
        float val = (which == 0) ? spike
                  : (which == 1) ? 0.5f * spike + 0.5f * x
                                 : x;
        out[(size_t)(m0 + m) * 768 + n0 + n] = (_Float16)val;
      }
    }
  }
}

// ---------------------------------------------------------------------------
// Per-head scores = (q . k^T)/8; block tile 128(i) x 64(j), wave tile 32x32.
// Stores f32 scores; t = (score>=1) tile reduced into row sums (xh) and
// col sums (xw) via per-tile atomics.
__global__ __launch_bounds__(256) void k_scores(
    const _Float16* __restrict__ qh, const _Float16* __restrict__ kh,
    float* __restrict__ scores, float* __restrict__ xh_sum,
    float* __restrict__ xw_sum) {
  const int bh = blockIdx.z;  // b*12 + h
  const int b = bh / 12, h = bh % 12;
  const int i0 = blockIdx.x * 128;
  const int j0 = blockIdx.y * 64;
  const int t  = threadIdx.x;
  const int wave = t >> 5;
  const int wm = wave >> 1, wn = wave & 1;

  __shared__ _Float16 Qs[128 * LDP];
  __shared__ _Float16 Ks[64 * LDP];
  __shared__ float Ts[128 * 65];

  v8f acc[2][2];
#pragma unroll
  for (int i = 0; i < 2; ++i)
#pragma unroll
    for (int j = 0; j < 2; ++j) acc[i][j] = (v8f){};

#pragma unroll
  for (int k0 = 0; k0 < 64; k0 += 32) {
    __syncthreads();
    {
      const int kk = t & 31, mm = t >> 5;
#pragma unroll
      for (int p = 0; p < 16; ++p) {
        const int row = mm + p * 8;
        Qs[row * LDP + kk] =
            qh[((size_t)b * 512 + i0 + row) * 768 + h * 64 + k0 + kk];
      }
#pragma unroll
      for (int p = 0; p < 8; ++p) {
        const int row = mm + p * 8;
        Ks[row * LDP + kk] =
            kh[((size_t)b * 512 + j0 + row) * 768 + h * 64 + k0 + kk];
      }
    }
    __syncthreads();
    v16h a[2], bb[2];
#pragma unroll
    for (int mi = 0; mi < 2; ++mi)
      a[mi] = frag_ld_a(Qs + (wm * 32 + mi * 16) * LDP);
#pragma unroll
    for (int ni = 0; ni < 2; ++ni)
      bb[ni] = frag_ld_b(Ks + (wn * 32 + ni * 16) * LDP);
#pragma unroll
    for (int mi = 0; mi < 2; ++mi)
#pragma unroll
      for (int ni = 0; ni < 2; ++ni)
        acc[mi][ni] = wmma32(a[mi], bb[ni], acc[mi][ni]);
  }

  const int lane = t & 31;
  const int nloc = lane & 15, mh = lane >> 4;
#pragma unroll
  for (int mi = 0; mi < 2; ++mi) {
#pragma unroll
    for (int ni = 0; ni < 2; ++ni) {
#pragma unroll
      for (int r = 0; r < 8; ++r) {
        const int m = wm * 32 + mi * 16 + r + 8 * mh;
        const int n = wn * 32 + ni * 16 + nloc;
        float sc = acc[mi][ni][r] * 0.125f;  // 1/sqrt(64)
        scores[((size_t)bh * 512 + i0 + m) * 512 + j0 + n] = sc;
        Ts[m * 65 + n] = (sc >= 1.f) ? 1.f : 0.f;
      }
    }
  }
  __syncthreads();
  if (t < 128) {
    float s = 0.f;
#pragma unroll 8
    for (int j = 0; j < 64; ++j) s += Ts[t * 65 + j];
    atomicAdd(&xh_sum[bh * 512 + i0 + t], s);
  } else if (t < 192) {
    const int c = t - 128;
    float s = 0.f;
#pragma unroll 8
    for (int r = 0; r < 128; ++r) s += Ts[r * 65 + c];
    atomicAdd(&xw_sum[bh * 512 + j0 + c], s);
  }
}

// ---------------------------------------------------------------------------
// Squeeze: y[b,r,n] = sum_c conv1_w[r,c] * cat[b,c,n]; accumulate BN stats.
__global__ __launch_bounds__(256) void k_squeeze(
    const float* __restrict__ xh_sum, const float* __restrict__ xw_sum,
    const float* __restrict__ conv1_w, float* __restrict__ y,
    float* __restrict__ bnstats /* [0..2]=sum, [4..6]=sumsq */) {
  const int gid = blockIdx.x * 256 + threadIdx.x;
  if (gid >= 8 * 1024) return;
  const int b = gid >> 10, n = gid & 1023;
  float cat[12];
#pragma unroll
  for (int c = 0; c < 12; ++c) {
    float s = (n < 512) ? xh_sum[((size_t)b * 12 + c) * 512 + n]
                        : xw_sum[((size_t)b * 12 + c) * 512 + (n - 512)];
    cat[c] = s * (1.f / 512.f);  // mean over S
  }
#pragma unroll
  for (int r = 0; r < 3; ++r) {
    float acc = 0.f;
#pragma unroll
    for (int c = 0; c < 12; ++c) acc += conv1_w[r * 12 + c] * cat[c];
    y[((size_t)b * 3 + r) * 1024 + n] = acc;
    atomicAdd(&bnstats[r], acc);
    atomicAdd(&bnstats[4 + r], acc * acc);
  }
}

// ---------------------------------------------------------------------------
// BN + relu + sigmoid gates: s_h[b,h,s], s_w[b,h,s].
__global__ __launch_bounds__(256) void k_gates(
    const float* __restrict__ y, const float* __restrict__ bnstats,
    const float* __restrict__ gamma, const float* __restrict__ beta,
    const float* __restrict__ convh_w, const float* __restrict__ convw_w,
    float* __restrict__ s_h, float* __restrict__ s_w) {
  const int gid = blockIdx.x * 256 + threadIdx.x;
  if (gid >= 8 * 12 * 512) return;
  const int b = gid / 6144, h = (gid / 512) % 12, s = gid & 511;
  float ah = 0.f, aw = 0.f;
#pragma unroll
  for (int r = 0; r < 3; ++r) {
    const float mu   = bnstats[r] * (1.f / 8192.f);
    const float var  = bnstats[4 + r] * (1.f / 8192.f) - mu * mu;
    const float rstd = rsqrtf(var + 1e-5f);
    float yh = y[((size_t)b * 3 + r) * 1024 + s];
    float yw = y[((size_t)b * 3 + r) * 1024 + 512 + s];
    yh = fmaxf((yh - mu) * rstd * gamma[r] + beta[r], 0.f);
    yw = fmaxf((yw - mu) * rstd * gamma[r] + beta[r], 0.f);
    ah += convh_w[h * 3 + r] * yh;
    aw += convw_w[h * 3 + r] * yw;
  }
  s_h[gid] = 1.f / (1.f + expf(-ah));
  s_w[gid] = 1.f / (1.f + expf(-aw));
}

// ---------------------------------------------------------------------------
// Softmax row kernel: bias = t * s_h[i] * s_w[j]; probs stored f16.
__global__ __launch_bounds__(256) void k_softmax(
    const float* __restrict__ scores, const float* __restrict__ mask,
    const float* __restrict__ s_h, const float* __restrict__ s_w,
    _Float16* __restrict__ probsh) {
  const int row = blockIdx.x;           // 0 .. B*H*S-1
  const int bh = row >> 9, i = row & 511, b = bh / 12;
  const float* src = scores + (size_t)row * 512;
  const float shi = s_h[bh * 512 + i];
  const int t = threadIdx.x;
  __shared__ float red[256];

  float v0, v1;
  {
    const float sc0 = src[t], sc1 = src[t + 256];
    const float t0 = (sc0 >= 1.f) ? 1.f : 0.f;
    const float t1 = (sc1 >= 1.f) ? 1.f : 0.f;
    v0 = sc0 + mask[b * 512 + t]       + t0 * shi * s_w[bh * 512 + t];
    v1 = sc1 + mask[b * 512 + t + 256] + t1 * shi * s_w[bh * 512 + t + 256];
  }
  red[t] = fmaxf(v0, v1);
  __syncthreads();
  for (int o = 128; o > 0; o >>= 1) {
    if (t < o) red[t] = fmaxf(red[t], red[t + o]);
    __syncthreads();
  }
  const float m = red[0];
  __syncthreads();
  const float e0 = expf(v0 - m), e1 = expf(v1 - m);
  red[t] = e0 + e1;
  __syncthreads();
  for (int o = 128; o > 0; o >>= 1) {
    if (t < o) red[t] += red[t + o];
    __syncthreads();
  }
  const float inv = 1.f / red[0];
  _Float16* dst = probsh + (size_t)row * 512;
  dst[t]       = (_Float16)(e0 * inv);
  dst[t + 256] = (_Float16)(e1 * inv);
}

// ---------------------------------------------------------------------------
// Context GEMM: out[b, i, h*64+d] = sum_j probs[bh,i,j] * v[b,j,h*64+d]
// Block tile 128(i) x 64(d), wave tile 32x32, K=512 in 16 steps.
__global__ __launch_bounds__(256) void k_ctx(
    const _Float16* __restrict__ probsh, const _Float16* __restrict__ vh,
    float* __restrict__ out) {
  const int bh = blockIdx.z;
  const int b = bh / 12, h = bh % 12;
  const int i0 = blockIdx.x * 128;
  const int t  = threadIdx.x;
  const int wave = t >> 5;
  const int wm = wave >> 1, wn = wave & 1;  // wn: d-half 0..31 / 32..63

  __shared__ _Float16 Ps[128 * LDP];
  __shared__ _Float16 Vs[64 * LDP];

  v8f acc[2][2];
#pragma unroll
  for (int i = 0; i < 2; ++i)
#pragma unroll
    for (int j = 0; j < 2; ++j) acc[i][j] = (v8f){};

  for (int k0 = 0; k0 < 512; k0 += 32) {
    __syncthreads();
    {  // stage probs tile 128x32
      const int kk = t & 31, mm = t >> 5;
      if (k0 + 32 < 512)
        __builtin_prefetch(&probsh[((size_t)bh * 512 + i0 + mm) * 512 + k0 + 32], 0, 1);
#pragma unroll
      for (int p = 0; p < 16; ++p) {
        const int row = mm + p * 8;
        Ps[row * LDP + kk] =
            probsh[((size_t)bh * 512 + i0 + row) * 512 + k0 + kk];
      }
    }
    {  // stage V tile transposed: Vs[d][j] = v[b, k0+j, h*64+d]
      const int dd = t & 63, kk0 = t >> 6;
#pragma unroll
      for (int p = 0; p < 8; ++p) {
        const int kk = kk0 + p * 4;
        Vs[dd * LDP + kk] =
            vh[((size_t)b * 512 + k0 + kk) * 768 + h * 64 + dd];
      }
    }
    __syncthreads();
    v16h a[2], bb[2];
#pragma unroll
    for (int mi = 0; mi < 2; ++mi)
      a[mi] = frag_ld_a(Ps + (wm * 32 + mi * 16) * LDP);
#pragma unroll
    for (int ni = 0; ni < 2; ++ni)
      bb[ni] = frag_ld_b(Vs + (wn * 32 + ni * 16) * LDP);
#pragma unroll
    for (int mi = 0; mi < 2; ++mi)
#pragma unroll
      for (int ni = 0; ni < 2; ++ni)
        acc[mi][ni] = wmma32(a[mi], bb[ni], acc[mi][ni]);
  }

  const int lane = t & 31;
  const int nloc = lane & 15, mh = lane >> 4;
#pragma unroll
  for (int mi = 0; mi < 2; ++mi) {
#pragma unroll
    for (int ni = 0; ni < 2; ++ni) {
#pragma unroll
      for (int r = 0; r < 8; ++r) {
        const int m = wm * 32 + mi * 16 + r + 8 * mh;
        const int n = wn * 32 + ni * 16 + nloc;
        out[((size_t)b * 512 + i0 + m) * 768 + h * 64 + n] = acc[mi][ni][r];
      }
    }
  }
}

// ---------------------------------------------------------------------------
extern "C" void kernel_launch(void* const* d_in, const int* in_sizes, int n_in,
                              void* d_out, int out_size, void* d_ws,
                              size_t ws_size, hipStream_t stream) {
  (void)in_sizes; (void)n_in; (void)out_size; (void)ws_size;
  const float* X       = (const float*)d_in[0];
  const float* mask    = (const float*)d_in[1];
  const float* Wq      = (const float*)d_in[2];
  const float* bq      = (const float*)d_in[3];
  const float* Wk      = (const float*)d_in[4];
  const float* bk      = (const float*)d_in[5];
  const float* Wv      = (const float*)d_in[6];
  const float* bv      = (const float*)d_in[7];
  const float* conv1_w = (const float*)d_in[8];
  const float* gamma   = (const float*)d_in[9];
  const float* beta    = (const float*)d_in[10];
  const float* convh_w = (const float*)d_in[11];
  const float* convw_w = (const float*)d_in[12];
  float* out = (float*)d_out;

  // workspace carving (256B aligned)
  size_t off = 0;
  auto carve = [&](size_t bytes) -> void* {
    void* p = (char*)d_ws + off;
    off += (bytes + 255) & ~(size_t)255;
    return p;
  };
  const size_t NQ = (size_t)4096 * 768;          // q/k/v halfs
  const size_t NS = (size_t)96 * 512 * 512;      // scores / probs elements
  _Float16* qh     = (_Float16*)carve(NQ * 2);
  _Float16* kh     = (_Float16*)carve(NQ * 2);
  _Float16* vh     = (_Float16*)carve(NQ * 2);
  float*    scores = (float*)   carve(NS * 4);
  _Float16* probsh = (_Float16*)carve(NS * 2);
  float*    xh_sum = (float*)   carve(49152 * 4);
  float*    xw_sum = (float*)   carve(49152 * 4);
  float*    ybuf   = (float*)   carve(8 * 3 * 1024 * 4);
  float*    bnst   = (float*)   carve(8 * 4);
  float*    s_h    = (float*)   carve(49152 * 4);
  float*    s_w    = (float*)   carve(49152 * 4);

  // 0) zero accumulators (ws is not re-zeroed between replays)
  k_zero<<<dim3((49152 + 255) / 256), dim3(256), 0, stream>>>(xh_sum, 49152);
  k_zero<<<dim3((49152 + 255) / 256), dim3(256), 0, stream>>>(xw_sum, 49152);
  k_zero<<<dim3(1), dim3(256), 0, stream>>>(bnst, 8);

  // 1) fused QKV projection + activations (WMMA, 128x128 block tiles)
  k_qkv<<<dim3(32, 6, 3), dim3(256), 0, stream>>>(X, Wq, Wk, Wv, bq, bk, bv,
                                                  qh, kh, vh);
  // 2) scores + t row/col sums (WMMA, 128x64 block tiles)
  k_scores<<<dim3(4, 8, 96), dim3(256), 0, stream>>>(qh, kh, scores, xh_sum,
                                                     xw_sum);
  // 3) squeeze conv + BN stats
  k_squeeze<<<dim3(32), dim3(256), 0, stream>>>(xh_sum, xw_sum, conv1_w, ybuf,
                                                bnst);
  // 4) BN + relu + sigmoid gates
  k_gates<<<dim3(192), dim3(256), 0, stream>>>(ybuf, bnst, gamma, beta,
                                               convh_w, convw_w, s_h, s_w);
  // 5) softmax with spiking bias
  k_softmax<<<dim3(49152), dim3(256), 0, stream>>>(scores, mask, s_h, s_w,
                                                   probsh);
  // 6) context GEMM (WMMA, 128x64 block tiles) -> final [B,S,768] output
  k_ctx<<<dim3(4, 1, 96), dim3(256), 0, stream>>>(probsh, vh, out);
}
